// GATClassifierLinear_85272280695082
// MI455X (gfx1250) — matrix operand order repaired and checked by reading, not compiled, not run on hardware
//
#include <hip/hip_runtime.h>
#include <hip/hip_bf16.h>

// ---- problem constants (fixed by the reference) ----
#define GAT_H       12
#define GAT_HID     32
#define GAT_HC      384
#define GAT_FIN     128
#define GAT_CLASSES 40
#define NEG_SLOPE   0.2f

// ---- WMMA vector types (CDNA5 / gfx1250, wave32) ----
typedef __attribute__((ext_vector_type(16))) __bf16 v16bf;
typedef __attribute__((ext_vector_type(8)))  float  v8f;
typedef __attribute__((ext_vector_type(4)))  int    v4i;

#if __has_builtin(__builtin_amdgcn_global_load_async_to_lds_b128)
#define HAVE_ASYNC_LDS 1
#else
#define HAVE_ASYNC_LDS 0
#endif

__device__ __forceinline__ unsigned short f32_to_bf16_rne(float f) {
    unsigned int b = __float_as_uint(f);
    unsigned int r = b + 0x7FFFu + ((b >> 16) & 1u);          // round-to-nearest-even
    if ((b & 0x7F800000u) == 0x7F800000u) r = b;              // inf/nan: truncate
    return (unsigned short)(r >> 16);
}

#if HAVE_ASYNC_LDS
// 16B per-lane async copy global -> LDS (tracked by ASYNCcnt)
__device__ __forceinline__ void async_cp16(const void* g, void* l) {
    __builtin_amdgcn_global_load_async_to_lds_b128(
        (__attribute__((address_space(1))) v4i*)g,
        (__attribute__((address_space(3))) v4i*)l, 0, 0);
}
__device__ __forceinline__ void wait_async_le0() {
#if __has_builtin(__builtin_amdgcn_s_wait_asynccnt)
    __builtin_amdgcn_s_wait_asynccnt(0);
#else
    asm volatile("s_wait_asynccnt 0x0" ::: "memory");
#endif
}
__device__ __forceinline__ void wait_async_le4() {
#if __has_builtin(__builtin_amdgcn_s_wait_asynccnt)
    __builtin_amdgcn_s_wait_asynccnt(4);
#else
    asm volatile("s_wait_asynccnt 0x4" ::: "memory");
#endif
}
#endif

// =====================================================================
// Elementwise: optional bias + optional ELU, emit bf16
// =====================================================================
__global__ __launch_bounds__(256)
void elu_cvt_kernel(const float* __restrict__ in, const float* __restrict__ bias, int C,
                    unsigned short* __restrict__ out, long n, int do_elu) {
    long idx = (long)blockIdx.x * blockDim.x + threadIdx.x;
    if (idx >= n) return;
    float v = in[idx];
    if (bias) v += bias[idx % C];
    if (do_elu) v = v > 0.f ? v : expm1f(v);
    out[idx] = f32_to_bf16_rne(v);
}

// =====================================================================
// Weight transpose + bf16 convert: W[KxN] (row major) -> WT[NxK] bf16
// =====================================================================
__global__ __launch_bounds__(256)
void transpose_cvt_kernel(const float* __restrict__ W, unsigned short* __restrict__ WT,
                          int K, int Nc) {
    int idx = blockIdx.x * blockDim.x + threadIdx.x;
    if (idx >= K * Nc) return;
    int k = idx / Nc, n = idx % Nc;
    WT[(size_t)n * K + k] = f32_to_bf16_rne(W[idx]);
}

// =====================================================================
// WMMA bf16 GEMM:  C[MxN] = A[MxK] * B[KxN]  (B supplied transposed: BT[NxK])
//   block = 256 threads (8 waves), tile 128x128, K-step 32
//   double-buffered LDS; staging via GLOBAL_LOAD_ASYNC_TO_LDS_B128 (ASYNCcnt)
//   A rows and BT rows must be padded/zero-filled to the tile grid (host does).
// =====================================================================
#define BM 128
#define BN 128
#define BK 32
#define LDSS 40   // u16 elements per LDS row (80B stride, 16B aligned, padded)

__global__ __launch_bounds__(256)
void wmma_gemm_bf16(const unsigned short* __restrict__ A,   // [M_pad x K] bf16 row-major
                    const unsigned short* __restrict__ BT,  // [N_pad x K] bf16 row-major (B^T)
                    const float* __restrict__ bias,         // [N] or nullptr
                    float* __restrict__ C,                  // [M x N] fp32
                    int M, int Nc, int K) {
    __shared__ unsigned short sA[2][BM * LDSS];
    __shared__ unsigned short sB[2][BN * LDSS];

    const int tid  = threadIdx.x;
    const int lane = tid & 31;
    const int wave = tid >> 5;
    const int wr   = wave >> 1;      // 0..3  (row group of 32)
    const int wc   = wave & 1;       // 0..1  (col group of 64)
    const int m0   = blockIdx.x * BM;
    const int n0   = blockIdx.y * BN;

    v8f acc[2][4];
#pragma unroll
    for (int i = 0; i < 2; i++)
#pragma unroll
        for (int j = 0; j < 4; j++)
#pragma unroll
            for (int l = 0; l < 8; l++) acc[i][j][l] = 0.f;

    const int lrow  = tid >> 1;            // 0..127 : tile row staged by this thread
    const int lhalf = (tid & 1) * 16;      // 0 or 16 : element offset within K-slab

    const unsigned short* gA = &A[(size_t)(m0 + lrow) * K + lhalf];
    const unsigned short* gB = &BT[(size_t)(n0 + lrow) * K + lhalf];
    const int sOff = lrow * LDSS + lhalf;

    auto stage = [&](int buf, int k0) {
#if HAVE_ASYNC_LDS
        async_cp16(gA + k0,     &sA[buf][sOff]);
        async_cp16(gA + k0 + 8, &sA[buf][sOff + 8]);
        async_cp16(gB + k0,     &sB[buf][sOff]);
        async_cp16(gB + k0 + 8, &sB[buf][sOff + 8]);
#else
        const uint4* pa = reinterpret_cast<const uint4*>(gA + k0);
        const uint4* pb = reinterpret_cast<const uint4*>(gB + k0);
        uint4 a0 = pa[0], a1 = pa[1];
        uint4 b0 = pb[0], b1 = pb[1];
        uint4* qa = reinterpret_cast<uint4*>(&sA[buf][sOff]);
        uint4* qb = reinterpret_cast<uint4*>(&sB[buf][sOff]);
        qa[0] = a0; qa[1] = a1;
        qb[0] = b0; qb[1] = b1;
#endif
    };

    const int rsel = lane & 15;
    const int kg   = lane >> 4;
    const int nk   = K / BK;

    stage(0, 0);

    for (int ki = 0; ki < nk; ki++) {
        const int buf = ki & 1;
        const bool has_next = (ki + 1) < nk;
        if (has_next) stage(buf ^ 1, (ki + 1) * BK);
#if HAVE_ASYNC_LDS
        if (has_next) wait_async_le4();    // current buffer's 4 copies complete
        else          wait_async_le0();
#endif
        __syncthreads();

        // ---- build fragments (ISA 7.12.2 16-bit layouts) ----
        union Frag { uint4 q[2]; v16bf v; } afr[2], bfr[4];
#pragma unroll
        for (int mi = 0; mi < 2; mi++) {
            // A 16x32: lanegroup0 holds K 0..7 & 16..23; group1 holds 8..15 & 24..31
            const unsigned short* base = &sA[buf][(wr * 32 + mi * 16 + rsel) * LDSS];
            afr[mi].q[0] = *reinterpret_cast<const uint4*>(base + kg * 8);
            afr[mi].q[1] = *reinterpret_cast<const uint4*>(base + 16 + kg * 8);
        }
#pragma unroll
        for (int ni = 0; ni < 4; ni++) {
            // B 32x16: lanegroup0 holds K 0..7 & 16..23; group1 holds 8..15 & 24..31
            const unsigned short* base = &sB[buf][(wc * 64 + ni * 16 + rsel) * LDSS];
            bfr[ni].q[0] = *reinterpret_cast<const uint4*>(base + kg * 8);
            bfr[ni].q[1] = *reinterpret_cast<const uint4*>(base + 16 + kg * 8);
        }

#pragma unroll
        for (int mi = 0; mi < 2; mi++)
#pragma unroll
            for (int ni = 0; ni < 4; ni++)
                acc[mi][ni] = __builtin_amdgcn_wmma_f32_16x16x32_bf16(
                    false, afr[mi].v, false, bfr[ni].v,
                    (short)0, acc[mi][ni], false, false);
        __syncthreads();
    }

    // ---- epilogue: C fragment layout (vgpr r: M = r + 8*(lane>=16), N = lane&15)
    const int rrow = (lane >> 4) * 8;
    const int ccol = lane & 15;
#pragma unroll
    for (int mi = 0; mi < 2; mi++) {
#pragma unroll
        for (int ni = 0; ni < 4; ni++) {
            int colg = n0 + wc * 64 + ni * 16 + ccol;
            if (colg >= Nc) continue;
            float badd = bias ? bias[colg] : 0.f;
#pragma unroll
            for (int r = 0; r < 8; r++) {
                int rowg = m0 + wr * 32 + mi * 16 + rrow + r;
                if (rowg < M) C[(size_t)rowg * Nc + colg] = acc[mi][ni][r] + badd;
            }
        }
    }
}

// =====================================================================
// Edge phase
// =====================================================================
__device__ __forceinline__ void edge_pair(const int* __restrict__ ei, int E, int e,
                                          int& s, int& d) {
    if (e < E) { s = ei[e]; d = ei[E + e]; }
    else       { s = d = e - E; }                 // self loops appended
}

// wave-per-edge: logits + fused segment-max via monotone-uint atomicMax
__global__ __launch_bounds__(256)
void edge_logits_kernel(const float* __restrict__ xl, const float* __restrict__ xr,
                        const float* __restrict__ att, const int* __restrict__ ei,
                        int E, int Nn, float* __restrict__ logit,
                        unsigned int* __restrict__ key) {
    int w    = (int)(((long)blockIdx.x * blockDim.x + threadIdx.x) >> 5);
    int lane = threadIdx.x & 31;
    int EE   = E + Nn;
    if (w >= EE) return;
    int s, d; edge_pair(ei, E, w, s, d);
    const float* ps = xl + (size_t)s * GAT_HC;
    const float* pd = xr + (size_t)d * GAT_HC;
#pragma unroll 1
    for (int h = 0; h < GAT_H; h++) {
        float v = ps[h * GAT_HID + lane] + pd[h * GAT_HID + lane];
        v = v > 0.f ? v : NEG_SLOPE * v;          // LeakyReLU
        v *= att[h * GAT_HID + lane];
#pragma unroll
        for (int off = 16; off; off >>= 1) v += __shfl_xor(v, off, 32);
        if (lane == 0) {
            logit[(size_t)w * GAT_H + h] = v;
            unsigned int b  = __float_as_uint(v);
            unsigned int kk = (b & 0x80000000u) ? ~b : (b | 0x80000000u);
            atomicMax(&key[(size_t)d * GAT_H + h], kk);
        }
    }
}

// thread-per-(edge,head): exp(logit - max) in place + segment-sum
__global__ __launch_bounds__(256)
void edge_exp_kernel(float* __restrict__ logit, const unsigned int* __restrict__ key,
                     float* __restrict__ denom, const int* __restrict__ ei,
                     int E, int Nn) {
    long idx = (long)blockIdx.x * blockDim.x + threadIdx.x;
    long EE  = (long)E + Nn;
    if (idx >= EE * GAT_H) return;
    int e = (int)(idx / GAT_H), h = (int)(idx % GAT_H);
    int d = (e < E) ? ei[E + e] : (e - E);
    unsigned int kk = key[(size_t)d * GAT_H + h];
    unsigned int b  = (kk & 0x80000000u) ? (kk ^ 0x80000000u) : ~kk;
    float m  = __uint_as_float(b);
    float ex = expf(logit[idx] - m);
    logit[idx] = ex;
    atomicAdd(&denom[(size_t)d * GAT_H + h], ex);
}

// wave-per-edge: alpha-weighted scatter-add of source messages
__global__ __launch_bounds__(256)
void edge_scatter_kernel(const float* __restrict__ xl, const float* __restrict__ ex,
                         const float* __restrict__ denom, const int* __restrict__ ei,
                         int E, int Nn, float* __restrict__ acc) {
    int w    = (int)(((long)blockIdx.x * blockDim.x + threadIdx.x) >> 5);
    int lane = threadIdx.x & 31;
    int EE   = E + Nn;
    if (w >= EE) return;
    int s, d; edge_pair(ei, E, w, s, d);
    const float* ps = xl + (size_t)s * GAT_HC;
    float* pd = acc + (size_t)d * GAT_HC;
#pragma unroll 1
    for (int h = 0; h < GAT_H; h++) {
        float alpha = ex[(size_t)w * GAT_H + h] / denom[(size_t)d * GAT_H + h];
        atomicAdd(&pd[h * GAT_HID + lane], alpha * ps[h * GAT_HID + lane]);
    }
}

// wave-per-row log_softmax over 40 classes, in place
__global__ __launch_bounds__(256)
void log_softmax_kernel(float* __restrict__ out, int Nn) {
    int w    = (int)(((long)blockIdx.x * blockDim.x + threadIdx.x) >> 5);
    int lane = threadIdx.x & 31;
    if (w >= Nn) return;
    float* row = out + (size_t)w * GAT_CLASSES;
    float v0 = lane < GAT_CLASSES ? row[lane] : -__builtin_inff();
    float v1 = (lane + 32) < GAT_CLASSES ? row[lane + 32] : -__builtin_inff();
    float m  = fmaxf(v0, v1);
#pragma unroll
    for (int off = 16; off; off >>= 1) m = fmaxf(m, __shfl_xor(m, off, 32));
    float s = (lane < GAT_CLASSES ? expf(v0 - m) : 0.f) +
              ((lane + 32) < GAT_CLASSES ? expf(v1 - m) : 0.f);
#pragma unroll
    for (int off = 16; off; off >>= 1) s += __shfl_xor(s, off, 32);
    float l = m + logf(s);
    if (lane < GAT_CLASSES)        row[lane]      = v0 - l;
    if (lane + 32 < GAT_CLASSES)   row[lane + 32] = v1 - l;
}

// =====================================================================
// Host orchestration
// =====================================================================
extern "C" void kernel_launch(void* const* d_in, const int* in_sizes, int n_in,
                              void* d_out, int out_size, void* d_ws, size_t ws_size,
                              hipStream_t stream) {
    const float* x   = (const float*)d_in[0];
    const int*   ei  = (const int*)d_in[1];
    const float* Wl1 = (const float*)d_in[2];
    const float* Wr1 = (const float*)d_in[3];
    const float* a1  = (const float*)d_in[4];
    const float* b1  = (const float*)d_in[5];
    const float* Wl2 = (const float*)d_in[6];
    const float* Wr2 = (const float*)d_in[7];
    const float* a2  = (const float*)d_in[8];
    const float* b2  = (const float*)d_in[9];
    const float* W3  = (const float*)d_in[10];
    const float* b3  = (const float*)d_in[11];
    const float* W4  = (const float*)d_in[12];
    const float* b4  = (const float*)d_in[13];
    (void)n_in; (void)out_size; (void)ws_size;

    const int Nn     = in_sizes[0] / GAT_FIN;
    const int E      = in_sizes[1] / 2;
    const int EE     = E + Nn;
    const int Nn_pad = ((Nn + BM - 1) / BM) * BM;   // A rows padded to tile grid

    // ---- workspace carve-up ----
    char*  ws  = (char*)d_ws;
    size_t off = 0;
    auto alloc = [&](size_t bytes) -> char* {
        char* p = ws + off;
        off += (bytes + 255) & ~(size_t)255;
        return p;
    };
    unsigned short* Xbf   = (unsigned short*)alloc((size_t)Nn_pad * GAT_HC * 2);  // x / h1 / h2 (bf16, padded)
    unsigned short* H3bf  = (unsigned short*)alloc((size_t)Nn_pad * GAT_HID * 2);
    unsigned short* WlT   = (unsigned short*)alloc((size_t)GAT_HC * GAT_HC * 2);
    unsigned short* WrT   = (unsigned short*)alloc((size_t)GAT_HC * GAT_HC * 2);
    float*          xl    = (float*)alloc((size_t)Nn * GAT_HC * 4);
    float*          xr    = (float*)alloc((size_t)Nn * GAT_HC * 4);
    float*          accb  = (float*)alloc((size_t)Nn * GAT_HC * 4);
    float*          exbuf = (float*)alloc((size_t)EE * GAT_H * 4);
    unsigned int*   key   = (unsigned int*)alloc((size_t)Nn * GAT_H * 4);
    float*          denom = (float*)alloc((size_t)Nn * GAT_H * 4);
    float*          h3f   = (float*)alloc((size_t)Nn * GAT_HID * 4);

    dim3 blk(256);
    auto cdiv = [](long a, long b) { return (int)((a + b - 1) / b); };

    // Zero padded tails so GEMM staging needs no guards.
    // Xbf tail covers pad rows for both K=128 (layer1) and K=384 (layer2/MLP):
    // valid layer-2 data [0, Nn*384*2) is rewritten later by elu_cvt.
    {
        size_t start = (size_t)Nn * GAT_FIN * 2;
        size_t end   = (size_t)Nn_pad * GAT_HC * 2;
        if (end > start) (void)hipMemsetAsync((char*)Xbf + start, 0, end - start, stream);
        size_t s3 = (size_t)Nn * GAT_HID * 2;
        size_t e3 = (size_t)Nn_pad * GAT_HID * 2;
        if (e3 > s3) (void)hipMemsetAsync((char*)H3bf + s3, 0, e3 - s3, stream);
    }

    // x -> bf16
    elu_cvt_kernel<<<cdiv((long)Nn * GAT_FIN, 256), blk, 0, stream>>>(
        x, nullptr, 1, Xbf, (long)Nn * GAT_FIN, 0);

    auto run_gat = [&](int Fin, const float* Wl, const float* Wr,
                       const float* att, const float* bias) {
        transpose_cvt_kernel<<<cdiv((long)Fin * GAT_HC, 256), blk, 0, stream>>>(Wl, WlT, Fin, GAT_HC);
        transpose_cvt_kernel<<<cdiv((long)Fin * GAT_HC, 256), blk, 0, stream>>>(Wr, WrT, Fin, GAT_HC);
        dim3 grid(cdiv(Nn, BM), cdiv(GAT_HC, BN));
        wmma_gemm_bf16<<<grid, blk, 0, stream>>>(Xbf, WlT, nullptr, xl, Nn, GAT_HC, Fin);
        wmma_gemm_bf16<<<grid, blk, 0, stream>>>(Xbf, WrT, nullptr, xr, Nn, GAT_HC, Fin);
        (void)hipMemsetAsync(key,   0, (size_t)Nn * GAT_H * 4, stream);
        (void)hipMemsetAsync(denom, 0, (size_t)Nn * GAT_H * 4, stream);
        (void)hipMemsetAsync(accb,  0, (size_t)Nn * GAT_HC * 4, stream);
        edge_logits_kernel<<<cdiv((long)EE * 32, 256), blk, 0, stream>>>(
            xl, xr, att, ei, E, Nn, exbuf, key);
        edge_exp_kernel<<<cdiv((long)EE * GAT_H, 256), blk, 0, stream>>>(
            exbuf, key, denom, ei, E, Nn);
        edge_scatter_kernel<<<cdiv((long)EE * 32, 256), blk, 0, stream>>>(
            xl, exbuf, denom, ei, E, Nn, accb);
        elu_cvt_kernel<<<cdiv((long)Nn * GAT_HC, 256), blk, 0, stream>>>(
            accb, bias, GAT_HC, Xbf, (long)Nn * GAT_HC, 1);
    };

    run_gat(GAT_FIN, Wl1, Wr1, a1, b1);   // layer 1 (elu'd h1 -> Xbf)
    run_gat(GAT_HC,  Wl2, Wr2, a2, b2);   // layer 2 (elu'd h2 -> Xbf)

    // MLP head: h3 = elu(h2 @ W3 + b3)   (zero WT so N-tile pad rows are clean)
    (void)hipMemsetAsync(WlT, 0, (size_t)GAT_HC * GAT_HC * 2, stream);
    transpose_cvt_kernel<<<cdiv((long)GAT_HC * GAT_HID, 256), blk, 0, stream>>>(W3, WlT, GAT_HC, GAT_HID);
    {
        dim3 grid(cdiv(Nn, BM), cdiv(GAT_HID, BN));
        wmma_gemm_bf16<<<grid, blk, 0, stream>>>(Xbf, WlT, b3, h3f, Nn, GAT_HID, GAT_HC);
    }
    elu_cvt_kernel<<<cdiv((long)Nn * GAT_HID, 256), blk, 0, stream>>>(
        h3f, nullptr, 1, H3bf, (long)Nn * GAT_HID, 1);

    // logits = h3 @ W4 + b4  -> d_out, then log_softmax in place
    (void)hipMemsetAsync(WlT, 0, (size_t)GAT_HC * GAT_HC * 2, stream);
    transpose_cvt_kernel<<<cdiv((long)GAT_HID * GAT_CLASSES, 256), blk, 0, stream>>>(W4, WlT, GAT_HID, GAT_CLASSES);
    {
        dim3 grid(cdiv(Nn, BM), cdiv(GAT_CLASSES, BN));
        wmma_gemm_bf16<<<grid, blk, 0, stream>>>(H3bf, WlT, b4, (float*)d_out, Nn, GAT_CLASSES, GAT_HID);
    }
    log_softmax_kernel<<<cdiv((long)Nn * 32, 256), blk, 0, stream>>>((float*)d_out, Nn);
}
